// TransformerLayerEdges_21620865368355
// MI455X (gfx1250) — matrix operand order repaired and checked by reading, not compiled, not run on hardware
//
#include <hip/hip_runtime.h>

// ---------------------------------------------------------------------------
// Graph-transformer layer with edge features for MI455X (gfx1250, wave32).
// All GEMMs use v_wmma_f32_16x16x32_f16 (f16 in, f32 accumulate).
// Edge tensors dominate -> fused kernels, f16 intermediates where the data
// is consumed as f16 anyway, pre-swizzled f16 weights (B-fragment layout),
// async global->LDS staging for raw f16 strips.
// ---------------------------------------------------------------------------

typedef __attribute__((ext_vector_type(16))) _Float16 v16h;
typedef __attribute__((ext_vector_type(8)))  float    v8f;

#define DEV __device__ __forceinline__

static constexpr int  B_    = 4;
static constexpr int  N_    = 256;
static constexpr int  H_    = 128;
static constexpr int  F_    = 512;
static constexpr int  NHEAD = 8;
static constexpr float LN_EPS = 1e-5f;
static constexpr float SCALE  = 0.25f;   // 1/sqrt(16)

DEV v8f vzero() {
  v8f z = {0.f, 0.f, 0.f, 0.f, 0.f, 0.f, 0.f, 0.f};
  return z;
}

DEV v8f wmma_f16(v16h a, v16h b, v8f c) {
  return __builtin_amdgcn_wmma_f32_16x16x32_f16(false, a, false, b,
                                                (short)0, c, false, false);
}

// A fragment (16x32 f16, row-major, lda halfs; LDS or global). Per ISA layout:
// lanes 0-15: row m, K = {0..7, 16..23}; lanes 16-31: row m, K = {8..15, 24..31}.
DEV v16h frag_a16(const _Float16* A, int lda, int m, int k0) {
  int lane = threadIdx.x & 31;
  int kb = k0 + ((lane >> 4) << 3);
  const _Float16* p = A + (long)m * lda + kb;
  union { v16h v; struct { uint4 lo; uint4 hi; } q; } r;
  r.q.lo = *(const uint4*)(p);
  r.q.hi = *(const uint4*)(p + 16);
  return r.v;
}

// B fragment from pre-swizzled weights: fragment (nt, ks) stores, per lane,
// the 16 halves it needs contiguously. Two 16B loads, 1KB/wave coalesced.
DEV v16h frag_b_sw(const _Float16* W, int KS, int nt, int ks) {
  int lane = threadIdx.x & 31;
  const _Float16* p = W + (((long)(nt * KS + ks) * 32 + lane) << 4);
  union { v16h v; struct { uint4 lo; uint4 hi; } q; } r;
  r.q.lo = *(const uint4*)(p);
  r.q.hi = *(const uint4*)(p + 8);
  return r.v;
}

// Stage rows of f32 into LDS as f16: float4 loads, 8B packed stores.
DEV void stage_f16v(const float* __restrict__ src, _Float16* __restrict__ dst, int n4) {
  const float4* s = (const float4*)src;
  for (int t = threadIdx.x; t < n4; t += blockDim.x) {
    float4 v = s[t];
    union { _Float16 h[4]; uint2 u; } r;
    r.h[0] = (_Float16)v.x; r.h[1] = (_Float16)v.y;
    r.h[2] = (_Float16)v.z; r.h[3] = (_Float16)v.w;
    *(uint2*)(dst + 4 * t) = r.u;
  }
}

// Stage raw f16 strip global->LDS via the CDNA5 async copy engine.
// Each lane copies 16B chunks; LDS address = low 32 bits of generic pointer.
DEV void stage_copy_async(const _Float16* __restrict__ gsrc,
                          _Float16* __restrict__ lds, int n16) {
  unsigned base = (unsigned)(unsigned long long)(const void*)lds;
  for (int t = threadIdx.x; t < n16; t += blockDim.x) {
    unsigned loff = base + t * 16;
    const void* g = (const char*)gsrc + (long)t * 16;
    asm volatile("global_load_async_to_lds_b128 %0, %1, off"
                 :: "v"(loff), "v"(g) : "memory");
  }
  asm volatile("s_wait_asynccnt 0x0" ::: "memory");
}

// Parallel LayerNorm stats: 8 threads/row, 16-element partials + shfl butterfly.
DEV void ln_stats(const float* __restrict__ sRow, float* __restrict__ sMean,
                  float* __restrict__ sRstd) {
  int t = threadIdx.x;
  int lr = t >> 3, seg = t & 7;
  const float* p = sRow + lr * H_ + seg * 16;
  float s1 = 0.f, s2 = 0.f;
#pragma unroll
  for (int c = 0; c < 16; ++c) { float v = p[c]; s1 += v; s2 += v * v; }
  s1 += __shfl_xor(s1, 1); s2 += __shfl_xor(s2, 1);
  s1 += __shfl_xor(s1, 2); s2 += __shfl_xor(s2, 2);
  s1 += __shfl_xor(s1, 4); s2 += __shfl_xor(s2, 4);
  if (seg == 0) {
    float m = s1 * (1.0f / H_);
    float var = s2 * (1.0f / H_) - m * m;
    sMean[lr] = m;
    sRstd[lr] = rsqrtf(var + LN_EPS);
  }
}

// --------------------------------------------------------------------------
// Weight f32 -> f16 with swizzle into B-fragment layout.
// --------------------------------------------------------------------------
__global__ void k_cvt_sw(const float* __restrict__ src, _Float16* __restrict__ dst,
                         int K, int Nc) {
  int t = blockIdx.x * blockDim.x + threadIdx.x;
  if (t >= K * Nc) return;
  int i    = t & 15;
  int lane = (t >> 4) & 31;
  int fk   = t >> 9;
  int KS   = K >> 5;
  int nt   = fk / KS;
  int ks   = fk - nt * KS;
  int n    = nt * 16 + (lane & 15);
  int k    = ks * 32 + ((lane >> 4) << 3) + i + ((i >= 8) ? 8 : 0);
  dst[t] = (_Float16)src[(long)k * Nc + n];
}

// Swizzle V (= K-projection, f32, layout [b,n,H]) into per-(b,h) B-fragment
// layout: 32 matrices of 256x16, 4096 halves each.
__global__ void k_swz_v(const float* __restrict__ Kh, _Float16* __restrict__ Vsw) {
  int t = blockIdx.x * blockDim.x + threadIdx.x;   // < 32*4096
  int i    = t & 15;
  int lane = (t >> 4) & 31;
  int ks   = (t >> 9) & 7;
  int mat  = t >> 12;
  int hh   = mat & 7, bb = mat >> 3;
  int n    = lane & 15;
  int k    = ks * 32 + ((lane >> 4) << 3) + i + ((i >= 8) ? 8 : 0);
  Vsw[t] = (_Float16)Kh[((long)(bb * N_ + k)) * H_ + hh * 16 + n];
}

// --------------------------------------------------------------------------
// Q/K projections (V == K per reference bug). 32 rows/block, 8 waves.
// --------------------------------------------------------------------------
__global__ void k_proj_qk(const float* __restrict__ x,
                          const _Float16* __restrict__ hWq,
                          const _Float16* __restrict__ hWk,
                          const float* __restrict__ bq, const float* __restrict__ bk,
                          float* __restrict__ Qh, float* __restrict__ Kh) {
  __shared__ __align__(16) _Float16 sA[32 * H_];
  long r0 = (long)blockIdx.x * 32;
  stage_f16v(x + r0 * H_, sA, 32 * H_ / 4);
  __syncthreads();
  int w = threadIdx.x >> 5, lane = threadIdx.x & 31;
  int n = lane & 15, hi = lane >> 4;
  int col = 16 * w + n;
  v8f accQ[2], accK[2];
  accQ[0] = vzero(); accQ[1] = vzero(); accK[0] = vzero(); accK[1] = vzero();
#pragma unroll
  for (int ks = 0; ks < 4; ++ks) {
    v16h bqf = frag_b_sw(hWq, 4, w, ks);
    v16h bkf = frag_b_sw(hWk, 4, w, ks);
#pragma unroll
    for (int rt = 0; rt < 2; ++rt) {
      v16h a = frag_a16(sA + rt * 16 * H_, H_, n, ks * 32);
      accQ[rt] = wmma_f16(a, bqf, accQ[rt]);
      accK[rt] = wmma_f16(a, bkf, accK[rt]);
    }
  }
  float bqv = bq[col], bkv = bk[col];
#pragma unroll
  for (int rt = 0; rt < 2; ++rt)
#pragma unroll
    for (int r = 0; r < 8; ++r) {
      long row = r0 + rt * 16 + r + 8 * hi;
      Qh[row * H_ + col] = accQ[rt][r] + bqv;
      Kh[row * H_ + col] = accK[rt][r] + bkv;
    }
}

// --------------------------------------------------------------------------
// Fused: Ee = e@We+be ; inter = Q_i * K_j * Ee * scale (stored f16) ;
// logits = sum_d inter. 32 rows/block (fixed b,i ; 32 consecutive j).
// --------------------------------------------------------------------------
__global__ void k_edge_inter(const float* __restrict__ e,
                             const _Float16* __restrict__ hWe,
                             const float* __restrict__ be,
                             const float* __restrict__ Qh, const float* __restrict__ Kh,
                             _Float16* __restrict__ interH, float* __restrict__ logits) {
  __shared__ __align__(16) _Float16 sA[32 * H_];
  long r0 = (long)blockIdx.x * 32;
  int bb = (int)(r0 >> 16);
  int i  = (int)((r0 >> 8) & 255);
  int j0 = (int)(r0 & 255);
  stage_f16v(e + r0 * H_, sA, 32 * H_ / 4);
  __syncthreads();
  int w = threadIdx.x >> 5, lane = threadIdx.x & 31;
  int n = lane & 15, hi = lane >> 4;
  int col = 16 * w + n;                    // head w, d = n
  v8f acc[2];
  acc[0] = vzero(); acc[1] = vzero();
#pragma unroll
  for (int ks = 0; ks < 4; ++ks) {
    v16h bf = frag_b_sw(hWe, 4, w, ks);
#pragma unroll
    for (int rt = 0; rt < 2; ++rt) {
      v16h a = frag_a16(sA + rt * 16 * H_, H_, n, ks * 32);
      acc[rt] = wmma_f16(a, bf, acc[rt]);
    }
  }
  float bev = be[col];
  float qv  = Qh[((long)(bb * N_ + i)) * H_ + col];
#pragma unroll
  for (int rt = 0; rt < 2; ++rt)
#pragma unroll
    for (int r = 0; r < 8; ++r) {
      int  j   = j0 + rt * 16 + r + 8 * hi;
      long row = r0 + rt * 16 + r + 8 * hi;
      float kv  = Kh[((long)(bb * N_ + j)) * H_ + col];
      float val = (acc[rt][r] + bev) * qv * kv * SCALE;
      interH[row * H_ + col] = (_Float16)val;
      float s = val;
      s += __shfl_xor(s, 1); s += __shfl_xor(s, 2);
      s += __shfl_xor(s, 4); s += __shfl_xor(s, 8);
      if (n == 0)
        logits[(((long)(bb * NHEAD + w) * N_ + i)) * N_ + j] = s;
    }
}

// --------------------------------------------------------------------------
// Row softmax over j (rows of 256), mask added; writes f16 scores.
// --------------------------------------------------------------------------
__global__ void k_softmax(const float* __restrict__ logits,
                          const float* __restrict__ mask,
                          _Float16* __restrict__ scoresH) {
  __shared__ float red[256];
  long row = blockIdx.x;
  int tid = threadIdx.x;
  long idx = row * 256 + tid;
  float v = logits[idx] + mask[idx];
  red[tid] = v; __syncthreads();
  for (int s = 128; s > 0; s >>= 1) {
    if (tid < s) red[tid] = fmaxf(red[tid], red[tid + s]);
    __syncthreads();
  }
  float mx = red[0]; __syncthreads();
  float ev = __expf(v - mx);
  red[tid] = ev; __syncthreads();
  for (int s = 128; s > 0; s >>= 1) {
    if (tid < s) red[tid] += red[tid + s];
    __syncthreads();
  }
  scoresH[idx] = (_Float16)(ev * (1.0f / red[0]));
}

// --------------------------------------------------------------------------
// Fused GEMM(128x128, swizzled W) + bias + residual(f32) + LayerNorm.
// AH: A-matrix is f16 (async raw-copy staging) else f32 (convert staging).
// OH: output stored f16 else f32. 32 rows/block.
// --------------------------------------------------------------------------
template <bool AH, bool OH>
__global__ void k_gemm_res_ln(const void* __restrict__ Ain_,
                              const _Float16* __restrict__ W,
                              const float* __restrict__ bias,
                              const float* __restrict__ resid,
                              const float* __restrict__ gam, const float* __restrict__ bet,
                              void* __restrict__ out_) {
  __shared__ __align__(16) _Float16 sA[32 * H_];
  __shared__ float sRow[32 * H_];
  __shared__ float sMean[32], sRstd[32];
  long r0 = (long)blockIdx.x * 32;
  if (AH) stage_copy_async((const _Float16*)Ain_ + r0 * H_, sA, 32 * H_ * 2 / 16);
  else    stage_f16v((const float*)Ain_ + r0 * H_, sA, 32 * H_ / 4);
  __syncthreads();
  int w = threadIdx.x >> 5, lane = threadIdx.x & 31;
  int n = lane & 15, hi = lane >> 4;
  int col = 16 * w + n;
  v8f acc[2];
  acc[0] = vzero(); acc[1] = vzero();
#pragma unroll
  for (int ks = 0; ks < 4; ++ks) {
    v16h bf = frag_b_sw(W, 4, w, ks);
#pragma unroll
    for (int rt = 0; rt < 2; ++rt) {
      v16h a = frag_a16(sA + rt * 16 * H_, H_, n, ks * 32);
      acc[rt] = wmma_f16(a, bf, acc[rt]);
    }
  }
  float bv = bias[col];
#pragma unroll
  for (int rt = 0; rt < 2; ++rt)
#pragma unroll
    for (int r = 0; r < 8; ++r) {
      int lr = rt * 16 + r + 8 * hi;
      sRow[lr * H_ + col] = acc[rt][r] + bv + resid[(r0 + lr) * H_ + col];
    }
  __syncthreads();
  ln_stats(sRow, sMean, sRstd);
  __syncthreads();
  for (int t = threadIdx.x; t < 32 * H_; t += blockDim.x) {
    int lr = t >> 7, c = t & 127;
    float res = (sRow[t] - sMean[lr]) * sRstd[lr] * gam[c] + bet[c];
    if (OH) ((_Float16*)out_)[(r0 + lr) * H_ + c] = (_Float16)res;
    else    ((float*)out_)[(r0 + lr) * H_ + c]    = res;
  }
}

// --------------------------------------------------------------------------
// att = scores @ V (V = K-projection, per reference bug). One 16x16 tile per
// wave, K=256. A = f16 scores (ISA-order frags), B = pre-swizzled f16 V.
// --------------------------------------------------------------------------
__global__ void k_attn_av(const _Float16* __restrict__ scoresH,
                          const _Float16* __restrict__ Vsw,
                          float* __restrict__ attO) {
  int gw = blockIdx.x * 8 + (threadIdx.x >> 5);
  int lane = threadIdx.x & 31, n = lane & 15, hi = lane >> 4;
  int it = gw & 15, bh = gw >> 4;
  int hh = bh & 7, bb = bh >> 3;
  int i0 = it * 16;
  const _Float16* Arow = scoresH + (((long)(bb * NHEAD + hh)) * N_ + i0) * N_;
  const _Float16* Vb   = Vsw + (long)(bb * NHEAD + hh) * 4096;
  v8f acc = vzero();
#pragma unroll
  for (int ks = 0; ks < 8; ++ks) {
    v16h a  = frag_a16(Arow, N_, n, ks * 32);
    v16h bf = frag_b_sw(Vb, 8, 0, ks);
    acc = wmma_f16(a, bf, acc);
  }
#pragma unroll
  for (int r = 0; r < 8; ++r) {
    int row = i0 + r + 8 * hi;
    attO[((long)(bb * N_ + row)) * H_ + hh * 16 + n] = acc[r];
  }
}

// --------------------------------------------------------------------------
// Fused FFN + residual + LayerNorm: out = LN(relu(X@W1+b1)@W2+b2 + X).
// IH: X is f16 (async staging, residual read from LDS) else f32.
// 32 rows/block; 32x512 f16 hidden strip lives in LDS (never hits HBM).
// --------------------------------------------------------------------------
template <bool IH>
__global__ void k_ffn_ln(const void* __restrict__ Xin_,
                         const _Float16* __restrict__ W1, const float* __restrict__ b1,
                         const _Float16* __restrict__ W2, const float* __restrict__ b2,
                         const float* __restrict__ gam, const float* __restrict__ bet,
                         float* __restrict__ out) {
  __shared__ __align__(16) _Float16 sA[32 * H_];     //  8 KB
  __shared__ __align__(16) _Float16 sHid[32 * F_];   // 32 KB
  __shared__ float sRow[32 * H_];                    // 16 KB
  __shared__ float sMean[32], sRstd[32];
  long r0 = (long)blockIdx.x * 32;
  if (IH) stage_copy_async((const _Float16*)Xin_ + r0 * H_, sA, 32 * H_ * 2 / 16);
  else    stage_f16v((const float*)Xin_ + r0 * H_, sA, 32 * H_ / 4);
  __syncthreads();
  int w = threadIdx.x >> 5, lane = threadIdx.x & 31;
  int n = lane & 15, hi = lane >> 4;

  // GEMM1: hidden = relu(X@W1 + b1), 32x512 into LDS (f16)
  for (int ct = 0; ct < 4; ++ct) {
    int nt  = w + 8 * ct;
    int col = nt * 16 + n;
    v8f acc[2];
    acc[0] = vzero(); acc[1] = vzero();
#pragma unroll
    for (int ks = 0; ks < 4; ++ks) {
      v16h bf = frag_b_sw(W1, 4, nt, ks);
#pragma unroll
      for (int rt = 0; rt < 2; ++rt) {
        v16h a = frag_a16(sA + rt * 16 * H_, H_, n, ks * 32);
        acc[rt] = wmma_f16(a, bf, acc[rt]);
      }
    }
    float bv = b1[col];
#pragma unroll
    for (int rt = 0; rt < 2; ++rt)
#pragma unroll
      for (int r = 0; r < 8; ++r) {
        int lr = rt * 16 + r + 8 * hi;
        float v = acc[rt][r] + bv;
        sHid[lr * F_ + col] = (_Float16)(v > 0.f ? v : 0.f);
      }
  }
  __syncthreads();

  // GEMM2: y = hidden@W2 + b2 + X  (K = 512)
  int col = 16 * w + n;
  v8f acc[2];
  acc[0] = vzero(); acc[1] = vzero();
  for (int ks = 0; ks < 16; ++ks) {
    v16h bf = frag_b_sw(W2, 16, w, ks);
#pragma unroll
    for (int rt = 0; rt < 2; ++rt) {
      v16h a = frag_a16(sHid + rt * 16 * F_, F_, n, ks * 32);
      acc[rt] = wmma_f16(a, bf, acc[rt]);
    }
  }
  float bv = b2[col];
#pragma unroll
  for (int rt = 0; rt < 2; ++rt)
#pragma unroll
    for (int r = 0; r < 8; ++r) {
      int lr = rt * 16 + r + 8 * hi;
      float rv = IH ? (float)sA[lr * H_ + col]
                    : ((const float*)Xin_)[(r0 + lr) * H_ + col];
      sRow[lr * H_ + col] = acc[rt][r] + bv + rv;
    }
  __syncthreads();
  ln_stats(sRow, sMean, sRstd);
  __syncthreads();
  for (int t = threadIdx.x; t < 32 * H_; t += blockDim.x) {
    int lr = t >> 7, c = t & 127;
    out[(r0 + lr) * H_ + c] = (sRow[t] - sMean[lr]) * sRstd[lr] * gam[c] + bet[c];
  }
}

// --------------------------------------------------------------------------
extern "C" void kernel_launch(void* const* d_in, const int* in_sizes, int n_in,
                              void* d_out, int out_size, void* d_ws, size_t ws_size,
                              hipStream_t stream) {
  const float* x    = (const float*)d_in[0];
  const float* e    = (const float*)d_in[1];
  const float* mask = (const float*)d_in[2];
  const float* Wq   = (const float*)d_in[3];  const float* bq   = (const float*)d_in[4];
  const float* Wk   = (const float*)d_in[5];  const float* bk   = (const float*)d_in[6];
  // d_in[7]=Wv, d_in[8]=bv intentionally unused (reference uses K-proj for V)
  const float* We   = (const float*)d_in[9];  const float* be   = (const float*)d_in[10];
  const float* Woh  = (const float*)d_in[11]; const float* boh  = (const float*)d_in[12];
  const float* Woe  = (const float*)d_in[13]; const float* boe  = (const float*)d_in[14];
  const float* Wf1x = (const float*)d_in[15]; const float* bf1x = (const float*)d_in[16];
  const float* Wf2x = (const float*)d_in[17]; const float* bf2x = (const float*)d_in[18];
  const float* Wf1e = (const float*)d_in[19]; const float* bf1e = (const float*)d_in[20];
  const float* Wf2e = (const float*)d_in[21]; const float* bf2e = (const float*)d_in[22];
  const float* g1x  = (const float*)d_in[23]; const float* b1x  = (const float*)d_in[24];
  const float* g1e  = (const float*)d_in[25]; const float* b1e  = (const float*)d_in[26];
  float* out = (float*)d_out;
  (void)in_sizes; (void)n_in; (void)out_size; (void)ws_size;

  char* ws = (char*)d_ws;
  size_t off = 0;
  auto walloc = [&](size_t bytes) -> char* {
    char* p = ws + off;
    off += (bytes + 1023) & ~size_t(1023);
    return p;
  };
  const int HH = H_ * H_;            // 16384
  const int HF = H_ * F_;            // 65536
  _Float16* hWq   = (_Float16*)walloc(HH * 2);
  _Float16* hWk   = (_Float16*)walloc(HH * 2);
  _Float16* hWe   = (_Float16*)walloc(HH * 2);
  _Float16* hWoh  = (_Float16*)walloc(HH * 2);
  _Float16* hWoe  = (_Float16*)walloc(HH * 2);
  _Float16* hWf1x = (_Float16*)walloc(HF * 2);
  _Float16* hWf2x = (_Float16*)walloc(HF * 2);
  _Float16* hWf1e = (_Float16*)walloc(HF * 2);
  _Float16* hWf2e = (_Float16*)walloc(HF * 2);
  const long nRowsX = (long)B_ * N_;               // 1024
  const long nRowsE = (long)B_ * N_ * N_;          // 262144
  const long nScore = (long)B_ * NHEAD * N_ * N_;  // 2M
  float*    Qh      = (float*)walloc(nRowsX * H_ * 4);
  float*    Kh      = (float*)walloc(nRowsX * H_ * 4);
  float*    attO    = (float*)walloc(nRowsX * H_ * 4);
  float*    x1      = (float*)walloc(nRowsX * H_ * 4);
  float*    logits  = (float*)walloc(nScore * 4);            // 8 MB
  _Float16* scoresH = (_Float16*)walloc(nScore * 2);         // 4 MB
  _Float16* Vsw     = (_Float16*)walloc(32 * 4096 * 2);      // 256 KB
  _Float16* interH  = (_Float16*)walloc(nRowsE * H_ * 2);    // 67 MB
  _Float16* e1H     = (_Float16*)walloc(nRowsE * H_ * 2);    // 67 MB

  // 0) weights -> f16, swizzled to B-fragment layout (L2-resident afterwards)
  k_cvt_sw<<<HH / 256, 256, 0, stream>>>(Wq,   hWq,   H_, H_);
  k_cvt_sw<<<HH / 256, 256, 0, stream>>>(Wk,   hWk,   H_, H_);
  k_cvt_sw<<<HH / 256, 256, 0, stream>>>(We,   hWe,   H_, H_);
  k_cvt_sw<<<HH / 256, 256, 0, stream>>>(Woh,  hWoh,  H_, H_);
  k_cvt_sw<<<HH / 256, 256, 0, stream>>>(Woe,  hWoe,  H_, H_);
  k_cvt_sw<<<HF / 256, 256, 0, stream>>>(Wf1x, hWf1x, H_, F_);
  k_cvt_sw<<<HF / 256, 256, 0, stream>>>(Wf2x, hWf2x, F_, H_);
  k_cvt_sw<<<HF / 256, 256, 0, stream>>>(Wf1e, hWf1e, H_, F_);
  k_cvt_sw<<<HF / 256, 256, 0, stream>>>(Wf2e, hWf2e, F_, H_);

  // 1) Q/K projections (V == K)
  k_proj_qk<<<nRowsX / 32, 256, 0, stream>>>(x, hWq, hWk, bq, bk, Qh, Kh);
  // 1b) swizzle V into B-fragment layout
  k_swz_v<<<32 * 4096 / 256, 256, 0, stream>>>(Kh, Vsw);
  // 2) fused edge projection + Q*K*Ee*scale (f16) + per-head logits
  k_edge_inter<<<nRowsE / 32, 256, 0, stream>>>(e, hWe, be, Qh, Kh, interH, logits);
  // 3) softmax over j -> f16 scores
  k_softmax<<<B_ * NHEAD * N_, 256, 0, stream>>>(logits, mask, scoresH);
  // 4) e1 = LN(inter@Woe + boe + e) -> f16
  k_gemm_res_ln<true, true><<<nRowsE / 32, 256, 0, stream>>>(
      interH, hWoe, boe, e, g1e, b1e, e1H);
  // 5) att = scores @ V
  k_attn_av<<<(B_ * NHEAD * (N_ / 16)) / 8, 256, 0, stream>>>(scoresH, Vsw, attO);
  // 6) x1 = LN(att@Woh + boh + x) -> f32
  k_gemm_res_ln<false, false><<<nRowsX / 32, 256, 0, stream>>>(
      attO, hWoh, boh, x, g1x, b1x, x1);
  // 7) x2 = LN(FFN(x1) + x1) -> out[0 : B*N*H)
  k_ffn_ln<false><<<nRowsX / 32, 256, 0, stream>>>(
      x1, hWf1x, bf1x, hWf2x, bf2x, g1x, b1x, out);
  // 8) e2 = LN(FFN(e1) + e1) -> out[B*N*H : ...)
  k_ffn_ln<true><<<nRowsE / 32, 256, 0, stream>>>(
      e1H, hWf1e, bf1e, hWf2e, bf2e, g1e, b1e, out + nRowsX * H_);
}
